// FutureEncoder_29240137351214
// MI455X (gfx1250) — compile-verified, matrix operand order checked by reading, not run on hardware
//
#include <hip/hip_runtime.h>
#include <hip/hip_bf16.h>

// ---------------------------------------------------------------------------
// FutureEncoder fused pipeline for gfx1250 (MI455X, wave32, WMMA).
//
// Algebraic folding: input_fc -> pos_enc concat -> input_fc2 -> input_fc3 is
// one affine map of the flattened 48-dim input per (b,n):
//     ftraj[bn] = x48[bn] @ W_comb[48,256] + bias_comb + W_fc3[256+agent]
// W_comb etc. are computed on-device (k_precompute, ~8M MACs, one block).
//
// All GEMMs use v_wmma_f32_16x16x32_f16 (f16 in, f32 accumulate).
// B matrices are pre-packed to the ISA fragment layout so each lane's B
// fragment is 32 contiguous bytes; A fragments are two 16B LDS loads.
//
// Workspace requirement: ~94 MB.
// ---------------------------------------------------------------------------

typedef _Float16 half_t;
typedef __attribute__((ext_vector_type(16))) _Float16 v16h;
typedef __attribute__((ext_vector_type(8)))  float    v8f;
typedef unsigned int u32x4 __attribute__((ext_vector_type(4)));

#define BB   4096
#define NN   11
#define LL   12
#define DD   256
#define BNR  (BB * NN)      // 45056 rows
#define PASTD 960

union Frag { v16h h; u32x4 q[2]; };

// k index inside a 16-bit A/B fragment: element e (0..15), lane-half hf (0..1)
// k = ((e&8)<<1) + (e&7) + hf*8   (per CDNA5 ISA 7.12.2, 16-bit A 16x32)

template<int KT>
__device__ __forceinline__ v8f wmma_tile(const half_t* A, int AS,
                                         const half_t* Bp, int NT,
                                         int mt, int nt, int lane) {
  v8f acc = {0.f, 0.f, 0.f, 0.f, 0.f, 0.f, 0.f, 0.f};
  const int m  = lane & 15;
  const int hf = lane >> 4;
  const half_t* a = A + (mt * 16 + m) * AS + hf * 8;
  const half_t* b = Bp + (nt * 32 + lane) * 16;
#pragma unroll
  for (int kt = 0; kt < KT; ++kt) {
    Frag fa, fb;
    fa.q[0] = *(const u32x4*)(a);        // k = k0 + hf*8 .. +7
    fa.q[1] = *(const u32x4*)(a + 16);   // k = k0 + 16 + hf*8 .. +7
    fb.q[0] = *(const u32x4*)(b);        // packed fragment, 32B contiguous
    fb.q[1] = *(const u32x4*)(b + 8);
    acc = __builtin_amdgcn_wmma_f32_16x16x32_f16(false, fa.h, false, fb.h,
                                                 (short)0, acc, false, false);
    a += 32;
    b += NT * 512;
  }
  return acc;
}

// --------------------------- weight folding --------------------------------
__global__ __launch_bounds__(256) void k_precompute(
    const float* __restrict__ W_in, const float* __restrict__ b_in,
    const float* __restrict__ W_pos, const float* __restrict__ b_pos,
    const float* __restrict__ W_fc2, const float* __restrict__ b_fc2,
    const float* __restrict__ W_fc3, const float* __restrict__ b_fc3,
    float* W_eff, float* biasL, float* W_big, float* bias0,
    float* W_comb, float* bias_comb) {
  const int tid = threadIdx.x;
  // W_eff[4][256] = W_in @ W_pos_top
  for (int i = tid; i < 4 * 256; i += 256) {
    int r = i >> 8, d = i & 255;
    float s = 0.f;
    for (int k = 0; k < 256; ++k) s += W_in[r * 256 + k] * W_pos[k * 256 + d];
    W_eff[i] = s;
  }
  // biasL[12][256] = pe @ W_pos_bot + b_pos + b_in @ W_pos_top
  const float cexp = -logf(10000.f) / 256.f;
  for (int i = tid; i < 12 * 256; i += 256) {
    int l = i >> 8, d = i & 255;
    float s = b_pos[d];
    for (int k = 0; k < 256; ++k) s += b_in[k] * W_pos[k * 256 + d];
    for (int k = 0; k < 256; ++k) {
      float w = expf(cexp * (float)(k & ~1));
      float pe = (k & 1) ? cosf((float)l * w) : sinf((float)l * w);
      s += pe * W_pos[(256 + k) * 256 + d];
    }
    biasL[i] = s;
  }
  __syncthreads();
  // W_big[48][256]: row r=l*4+i : sum_k W_eff[i][k] * W_fc2[l*256+k][d]
  for (int i = tid; i < 48 * 256; i += 256) {
    int r = i >> 8, d = i & 255;
    int l = r >> 2, ii = r & 3;
    float s = 0.f;
    for (int k = 0; k < 256; ++k)
      s += W_eff[ii * 256 + k] * W_fc2[(l * 256 + k) * 256 + d];
    W_big[i] = s;
  }
  // bias0[256] = biasL_flat @ W_fc2 + b_fc2
  for (int i = tid; i < 256; i += 256) {
    float s = b_fc2[i];
    for (int lk = 0; lk < 3072; ++lk) s += biasL[lk] * W_fc2[lk * 256 + i];
    bias0[i] = s;
  }
  __syncthreads();
  // W_comb = W_big @ W_fc3_top ; bias_comb = bias0 @ W_fc3_top + b_fc3
  for (int i = tid; i < 48 * 256; i += 256) {
    int r = i >> 8, d = i & 255;
    float s = 0.f;
    for (int k = 0; k < 256; ++k) s += W_big[r * 256 + k] * W_fc3[k * 256 + d];
    W_comb[i] = s;
  }
  for (int i = tid; i < 256; i += 256) {
    float s = b_fc3[i];
    for (int k = 0; k < 256; ++k) s += bias0[k] * W_fc3[k * 256 + i];
    bias_comb[i] = s;
  }
}

// ------------------- pack f32 [K][N] -> WMMA B fragments -------------------
__global__ __launch_bounds__(256) void k_pack(const float* __restrict__ W,
                                              int K, int N, int KT, int NT,
                                              half_t* __restrict__ out) {
  int tile = blockIdx.x * (blockDim.x >> 5) + (threadIdx.x >> 5);
  if (tile >= KT * NT) return;
  int lane = threadIdx.x & 31;
  int kt = tile / NT, nt = tile % NT;
  int m = lane & 15, hf = lane >> 4;
  half_t* o = out + (tile * 32 + lane) * 16;
#pragma unroll
  for (int e = 0; e < 16; ++e) {
    int k = kt * 32 + ((e & 8) << 1) + (e & 7) + hf * 8;
    float v = (k < K) ? W[k * N + nt * 16 + m] : 0.0f;
    o[e] = (half_t)v;
  }
}

// --------------------------- stage 1: ftraj --------------------------------
__global__ __launch_bounds__(128) void k_ftraj(
    const float* __restrict__ X, const half_t* __restrict__ Wcp,
    const float* __restrict__ bias_comb, const float* __restrict__ W_fc3,
    float* __restrict__ ftraj) {
  __shared__ half_t Xs[32 * 72];             // 32 rows, K padded to 64, +8 pad
  const int row0 = blockIdx.x * 32;
  const int tid = threadIdx.x;
  for (int i = tid; i < 32 * 64; i += 128) {
    int r = i >> 6, k = i & 63;
    Xs[r * 72 + k] = (half_t)(k < 48 ? X[(row0 + r) * 48 + k] : 0.0f);
  }
  __syncthreads();
  const int lane = tid & 31, wave = tid >> 5;
  const int m = lane & 15, hf = lane >> 4;
  for (int t = wave; t < 32; t += 4) {
    int mt = t >> 4, nt = t & 15;
    v8f acc = wmma_tile<2>(Xs, 72, Wcp, 16, mt, nt, lane);
    int n = nt * 16 + m;
    float bn = bias_comb[n];
#pragma unroll
    for (int v = 0; v < 8; ++v) {
      int r = row0 + mt * 16 + v + hf * 8;
      int agent = r % NN;
      ftraj[r * 256 + n] = acc[v] + bn + W_fc3[(256 + agent) * 256 + n];
    }
  }
}

// ----------------- stage 2: per-scene graph ops (N = 11) -------------------
__global__ __launch_bounds__(128) void k_scene(
    const float* __restrict__ ftraj,
    half_t* __restrict__ agg_h, half_t* __restrict__ hag_h) {
  __shared__ float F[NN][260];
  __shared__ float S[NN][NN];
  __shared__ float attn[NN][NN];
  __shared__ float An[NN][NN];
  __shared__ float thr_s;
  const int b = blockIdx.x;
  const int tid = threadIdx.x;
  for (int i = tid; i < NN * 256; i += 128)
    F[i >> 8][i & 255] = ftraj[(b * NN) * 256 + i];
  __syncthreads();
  if (tid < NN * NN) {
    int n = tid / NN, mm = tid % NN;
    float s = 0.f;
    for (int k = 0; k < 256; ++k) s += F[n][k] * F[mm][k];
    S[n][mm] = s;
  }
  __syncthreads();
  if (tid == 0) {
    float a = 1e30f;
    for (int n = 0; n < NN; ++n)
      for (int mm = 0; mm < NN; ++mm) {
        float c = S[n][mm] * rsqrtf(S[n][n] * S[mm][mm]);
        a = fminf(a, c);
      }
    thr_s = (a < 0.4f) ? 0.4f : ((a > 0.4f && a < 0.6f) ? a + 0.1f : a + 0.03f);
  }
  __syncthreads();
  if (tid < NN) {
    int n = tid;
    float mx = -1e30f;
    for (int mm = 0; mm < NN; ++mm) mx = fmaxf(mx, S[n][mm] * 0.0625f);
    float sum = 0.f;
    for (int mm = 0; mm < NN; ++mm) {
      float e = expf(S[n][mm] * 0.0625f - mx);
      attn[n][mm] = e; sum += e;
    }
    float inv = 1.f / sum;
    for (int mm = 0; mm < NN; ++mm) attn[n][mm] *= inv;
    float cnt = 0.f;
    for (int mm = 0; mm < NN; ++mm) {
      float c = S[n][mm] * rsqrtf(S[n][n] * S[mm][mm]);
      float v = (c < thr_s) ? 0.f : 1.f;
      An[n][mm] = v; cnt += v;
    }
    float ninv = 1.f / fmaxf(cnt, 1.f);
    for (int mm = 0; mm < NN; ++mm) An[n][mm] *= ninv;
  }
  __syncthreads();
  for (int i = tid; i < NN * 256; i += 128) {
    int n = i >> 8, d = i & 255;
    float s1 = 0.f, s2 = 0.f;
    for (int mm = 0; mm < NN; ++mm) {
      float f = F[mm][d];
      s1 += attn[n][mm] * f;
      s2 += An[n][mm] * f;
    }
    agg_h[(b * NN + n) * 256 + d] = (half_t)s1;
    hag_h[(b * NN + n) * 256 + d] = (half_t)s2;
  }
}

// -------- stage 3: msg/hyp/line/out/qz fully fused in LDS (32-row tile) ----
#define AS_A 520     // [32][512] + 8 pad
#define AS_H 1544    // [32][1536] + 8 pad
#define AS_T 264     // [32][256] + 8 pad
#define AS_h 136     // [32][128] + 8 pad
#define OFF_H 16640  // halves
#define OFF_T 66048  // halves

__global__ __launch_bounds__(128) void k_fused(
    const float* __restrict__ ftraj, const half_t* __restrict__ agg_h,
    const half_t* __restrict__ hag_h, const float* __restrict__ past,
    const half_t* __restrict__ Wmsg_p, const float* __restrict__ b_msg,
    const half_t* __restrict__ Whyp_p, const float* __restrict__ b_hyp,
    const half_t* __restrict__ Wline_p,
    const half_t* __restrict__ Wout_p, const float* __restrict__ b_out,
    const half_t* __restrict__ Wqz_p, const float* __restrict__ b_qz,
    float* __restrict__ out) {
  extern __shared__ half_t sm[];
  half_t* bufA = sm;            // [32][520]  : [ftraj | agg/hag]
  half_t* bufH = sm + OFF_H;    // [32][1544] : [past | ftraj | inter | feat]
  half_t* bufT = sm + OFF_T;    // [32][264]  : hyp, later h [32][136]
  const int row0 = blockIdx.x * 32;
  const int tid = threadIdx.x;
  const int lane = tid & 31, wave = tid >> 5;
  const int m = lane & 15, hf = lane >> 4;

  for (int i = tid; i < 32 * 256; i += 128) {
    int r = i >> 8, c = i & 255;
    float f = ftraj[(row0 + r) * 256 + c];
    half_t h = (half_t)f;
    bufA[r * AS_A + c] = h;
    bufH[r * AS_H + 960 + c] = h;
    bufA[r * AS_A + 256 + c] = agg_h[(row0 + r) * 256 + c];
  }
  for (int i = tid; i < 32 * PASTD; i += 128) {
    int r = i / PASTD, c = i % PASTD;
    bufH[r * AS_H + c] = (half_t)past[(row0 + r) * PASTD + c];
  }
  __syncthreads();

  // GEMM1: inter = relu([ftraj|agg] @ W_msg + b_msg) -> bufH[:,1216:1472]
  for (int t = wave; t < 32; t += 4) {
    int mt = t >> 4, nt = t & 15;
    v8f acc = wmma_tile<16>(bufA, AS_A, Wmsg_p, 16, mt, nt, lane);
    float bn = b_msg[nt * 16 + m];
#pragma unroll
    for (int v = 0; v < 8; ++v) {
      int r = mt * 16 + v + hf * 8;
      bufH[r * AS_H + 1216 + nt * 16 + m] = (half_t)fmaxf(acc[v] + bn, 0.f);
    }
  }
  __syncthreads();
  for (int i = tid; i < 32 * 256; i += 128) {
    int r = i >> 8, c = i & 255;
    bufA[r * AS_A + 256 + c] = hag_h[(row0 + r) * 256 + c];
  }
  __syncthreads();
  // GEMM2: hyp = relu([ftraj|hag] @ W_hyp + b_hyp) -> bufT
  for (int t = wave; t < 32; t += 4) {
    int mt = t >> 4, nt = t & 15;
    v8f acc = wmma_tile<16>(bufA, AS_A, Whyp_p, 16, mt, nt, lane);
    float bn = b_hyp[nt * 16 + m];
#pragma unroll
    for (int v = 0; v < 8; ++v) {
      int r = mt * 16 + v + hf * 8;
      bufT[r * AS_T + nt * 16 + m] = (half_t)fmaxf(acc[v] + bn, 0.f);
    }
  }
  __syncthreads();
  // GEMM3: feat = hyp @ W_line -> bufH[:,1472:1536]
  for (int t = wave; t < 8; t += 4) {
    int mt = t >> 2, nt = t & 3;
    v8f acc = wmma_tile<8>(bufT, AS_T, Wline_p, 4, mt, nt, lane);
#pragma unroll
    for (int v = 0; v < 8; ++v) {
      int r = mt * 16 + v + hf * 8;
      bufH[r * AS_H + 1472 + nt * 16 + m] = (half_t)acc[v];
    }
  }
  __syncthreads();
  // GEMM4: h = relu(H @ W_out + b_out) -> bufT (stride 136)
  for (int t = wave; t < 16; t += 4) {
    int mt = t >> 3, nt = t & 7;
    v8f acc = wmma_tile<48>(bufH, AS_H, Wout_p, 8, mt, nt, lane);
    float bn = b_out[nt * 16 + m];
#pragma unroll
    for (int v = 0; v < 8; ++v) {
      int r = mt * 16 + v + hf * 8;
      bufT[r * AS_h + nt * 16 + m] = (half_t)fmaxf(acc[v] + bn, 0.f);
    }
  }
  __syncthreads();
  // GEMM5: out = h @ W_qz + b_qz -> global
  for (int t = wave; t < 8; t += 4) {
    int mt = t >> 2, nt = t & 3;
    v8f acc = wmma_tile<4>(bufT, AS_h, Wqz_p, 4, mt, nt, lane);
    float bn = b_qz[nt * 16 + m];
#pragma unroll
    for (int v = 0; v < 8; ++v) {
      int r = row0 + mt * 16 + v + hf * 8;
      out[r * 64 + nt * 16 + m] = acc[v] + bn;
    }
  }
}

// ---------------------------------------------------------------------------
extern "C" void kernel_launch(void* const* d_in, const int* in_sizes, int n_in,
                              void* d_out, int out_size, void* d_ws, size_t ws_size,
                              hipStream_t stream) {
  (void)in_sizes; (void)n_in; (void)out_size; (void)ws_size;
  const float* inputs = (const float*)d_in[0];
  const float* past   = (const float*)d_in[1];
  const float* W_in   = (const float*)d_in[2];
  const float* b_in   = (const float*)d_in[3];
  const float* W_pos  = (const float*)d_in[4];
  const float* b_pos  = (const float*)d_in[5];
  const float* W_fc2  = (const float*)d_in[6];
  const float* b_fc2  = (const float*)d_in[7];
  const float* W_fc3  = (const float*)d_in[8];
  const float* b_fc3  = (const float*)d_in[9];
  const float* W_msg  = (const float*)d_in[10];
  const float* b_msg  = (const float*)d_in[11];
  const float* W_hyp  = (const float*)d_in[12];
  const float* b_hyp  = (const float*)d_in[13];
  const float* W_line = (const float*)d_in[14];
  const float* W_out  = (const float*)d_in[15];
  const float* b_out  = (const float*)d_in[16];
  const float* W_qz   = (const float*)d_in[17];
  const float* b_qz   = (const float*)d_in[18];
  float* out = (float*)d_out;
  char* ws = (char*)d_ws;

  // workspace layout (all offsets 256B aligned)
  float*  W_eff     = (float*)(ws + 0);            //   4*256 f32
  float*  biasL     = (float*)(ws + 4096);         //  12*256 f32
  float*  W_big     = (float*)(ws + 16384);        //  48*256 f32
  float*  bias0     = (float*)(ws + 65536);        //     256 f32
  float*  W_comb    = (float*)(ws + 66560);        //  48*256 f32
  float*  bias_comb = (float*)(ws + 115712);       //     256 f32
  half_t* Wcomb_p   = (half_t*)(ws + 116736);      //  2*16 tiles
  half_t* Wmsg_p    = (half_t*)(ws + 149504);      // 16*16 tiles
  half_t* Whyp_p    = (half_t*)(ws + 411648);      // 16*16 tiles
  half_t* Wline_p   = (half_t*)(ws + 673792);      //  8*4  tiles
  half_t* Wout_p    = (half_t*)(ws + 706560);      // 48*8  tiles
  half_t* Wqz_p     = (half_t*)(ws + 1099776);     //  4*4  tiles
  float*  ftraj     = (float*)(ws + 1116160);      // BN*256 f32
  half_t* agg_h     = (half_t*)(ws + 47253504);    // BN*256 f16
  half_t* hag_h     = (half_t*)(ws + 70322176);    // BN*256 f16
  // total required: 93,390,848 bytes

  k_precompute<<<1, 256, 0, stream>>>(W_in, b_in, W_pos, b_pos, W_fc2, b_fc2,
                                      W_fc3, b_fc3, W_eff, biasL, W_big, bias0,
                                      W_comb, bias_comb);

  k_pack<<<4,  256, 0, stream>>>(W_comb, 48,  256, 2,  16, Wcomb_p);
  k_pack<<<32, 256, 0, stream>>>(W_msg, 512,  256, 16, 16, Wmsg_p);
  k_pack<<<32, 256, 0, stream>>>(W_hyp, 512,  256, 16, 16, Whyp_p);
  k_pack<<<4,  256, 0, stream>>>(W_line, 256, 64,  8,  4,  Wline_p);
  k_pack<<<48, 256, 0, stream>>>(W_out, 1536, 128, 48, 8,  Wout_p);
  k_pack<<<2,  256, 0, stream>>>(W_qz,  128,  64,  4,  4,  Wqz_p);

  k_ftraj<<<BNR / 32, 128, 0, stream>>>(inputs, Wcomb_p, bias_comb, W_fc3, ftraj);

  k_scene<<<BB, 128, 0, stream>>>(ftraj, agg_h, hag_h);

  size_t smem = (size_t)(OFF_T + 32 * AS_T) * sizeof(half_t); // 148,992 B
  k_fused<<<BNR / 32, 128, smem, stream>>>(ftraj, agg_h, hag_h, past,
                                           Wmsg_p, b_msg, Whyp_p, b_hyp,
                                           Wline_p, Wout_p, b_out,
                                           Wqz_p, b_qz, out);
}
